// Attention_41601053229865
// MI455X (gfx1250) — matrix operand order repaired and checked by reading, not compiled, not run on hardware
//
#include <hip/hip_runtime.h>
#include <hip/hip_bf16.h>
#include <math.h>

// ---------------------------------------------------------------------------
// Channel attention (XCA-style) for B=2, DIM=192, HEADS=4, 32^3 volume.
// GEMM stages: v_wmma_f32_16x16x32_f16 (wave32 CDNA5 WMMA).
// Score GEMM staging: Tensor Data Mover (tensor_load_to_lds + s_wait_tensorcnt)
// with a manual b128-copy fallback if the builtin is unavailable.
// ---------------------------------------------------------------------------

typedef __attribute__((ext_vector_type(16))) _Float16 v16h;
typedef __attribute__((ext_vector_type(8)))  _Float16 v8h;
typedef __attribute__((ext_vector_type(2)))  _Float16 h2;
typedef __attribute__((ext_vector_type(8)))  float    v8f;

#define SPATIAL 32768   // 32*32*32
#define NB      2       // batch
#define CDIM    192
#define NHEADS  4
#define CH      48      // CDIM / NHEADS

#if defined(__has_builtin)
#  if __has_builtin(__builtin_amdgcn_tensor_load_to_lds)
#    define HAVE_TDM 1
#  endif
#endif
#ifndef HAVE_TDM
#  define HAVE_TDM 0
#endif

__device__ __forceinline__ v16h cat8(v8h lo, v8h hi) {
  return __builtin_shufflevector(lo, hi, 0, 1, 2, 3, 4, 5, 6, 7,
                                 8, 9, 10, 11, 12, 13, 14, 15);
}

// A fragment (16x32 f16), tile row-major [m][ld]; two ds_load_b128 per lane.
__device__ __forceinline__ v16h frag_a(const _Float16* t, int ld, int lane) {
  const int m  = lane & 15;
  const int kb = (lane >> 4) << 3;
  const _Float16* base = t + m * ld;
  const v8h lo = *(const v8h*)(base + kb);
  const v8h hi = *(const v8h*)(base + 16 + kb);
  return cat8(lo, hi);
}

// B fragment (32x16 f16), tile COLUMN-major [n][ldk]; two ds_load_b128.
__device__ __forceinline__ v16h frag_b(const _Float16* t, int ldk, int lane) {
  const int n  = lane & 15;
  const int kb = (lane >> 4) << 3;
  const _Float16* base = t + n * ldk;
  const v8h lo = *(const v8h*)(base + kb);
  const v8h hi = *(const v8h*)(base + 16 + kb);
  return cat8(lo, hi);
}

__device__ __forceinline__ v8f wmma_f16(v16h a, v16h b, v8f c) {
  return __builtin_amdgcn_wmma_f32_16x16x32_f16(false, a, false, b,
                                                (short)0, c, false, false);
}

#if HAVE_TDM
typedef __attribute__((ext_vector_type(4))) unsigned int u32x4;
typedef __attribute__((ext_vector_type(8))) int          i32x8;
typedef __attribute__((ext_vector_type(4))) int          i32x4;

// Issue one TDM load of a 48-row x 32-halfword 2D tile (row stride SPATIAL
// halves) from global f16 into LDS (row-major, packed).  One instruction per
// wave; completion tracked with TENSORcnt.
__device__ __forceinline__ void tdm_load_48x32_f16(unsigned lds_addr,
                                                   const _Float16* gsrc) {
  const unsigned long long ga = (unsigned long long)(uintptr_t)gsrc;
  u32x4 g0;
  g0[0] = 1u;                                              // count=1
  g0[1] = lds_addr;                                        // LDS byte address
  g0[2] = (unsigned)(ga & 0xFFFFFFFFull);                  // global_addr lo
  g0[3] = (unsigned)((ga >> 32) & 0x01FFFFFFull) | (2u << 30); // hi | type=2
  i32x8 g1;
  g1[0] = 0x00010000;      // workgroup_mask=0, data_size=1 (2 bytes)
  g1[1] = 32 << 16;        // tensor_dim0 = 32
  g1[2] = 48 << 16;        // tensor_dim1 = 48
  g1[3] = 32 << 16;        // tile_dim0 = 32
  g1[4] = 48;              // tile_dim1 = 48, tile_dim2 = 0
  g1[5] = 32768;           // tensor_dim0_stride = SPATIAL (lo)
  g1[6] = 0;               // stride hi / dim1 stride lo
  g1[7] = 0;
  const i32x4 z4 = {0, 0, 0, 0};
  const i32x8 z8 = {0, 0, 0, 0, 0, 0, 0, 0};
  __builtin_amdgcn_tensor_load_to_lds(g0, g1, z4, z4, z8, 0);
}
#endif

// Stage a 16x32 weight tile (row-major, K contiguous in global) via float2.
__device__ __forceinline__ void stage_w16x32(_Float16* dst, const float* __restrict__ w,
                                             int o0, int C, int k0, int tid) {
  for (int i = tid; i < 256; i += 128) {
    const int r = i >> 4, kp = (i & 15) << 1;
    const float2 v = *(const float2*)(w + (size_t)(o0 + r) * C + (k0 + kp));
    h2 p; p[0] = (_Float16)v.x; p[1] = (_Float16)v.y;
    *(h2*)(dst + r * 32 + kp) = p;
  }
}

// Stage a 32(k) x 32(n) activation tile into column-major [n][32].
template <typename T>
__device__ __forceinline__ void stage_x32x32(_Float16* dst, const T* __restrict__ src,
                                             int lane) {
#pragma unroll
  for (int j = 0; j < 16; ++j) {
    const _Float16 a0 = (_Float16)src[(size_t)(2 * j) * SPATIAL + lane];
    const _Float16 a1 = (_Float16)src[(size_t)(2 * j + 1) * SPATIAL + lane];
    h2 p; p[0] = a0; p[1] = a1;
    *(h2*)(dst + lane * 32 + 2 * j) = p;
  }
}

// ---------------------------------------------------------------------------
// 1x1x1 conv as GEMM: out[b,o,n] = sum_c W[o,c] * in[b,c,n] + bias[o]
// Block = 4 waves; each wave owns a 16x32 output tile (2 WMMAs per A frag).
// Double-buffered K pipeline: one barrier per K step.
// ---------------------------------------------------------------------------
template <typename T>
__global__ __launch_bounds__(128)
void conv1x1_wmma(const T* __restrict__ in, const float* __restrict__ w,
                  const float* __restrict__ bias, float* __restrict__ out,
                  int O, int C) {
  __shared__ alignas(16) _Float16 As[2][16 * 32];        // weight tile
  __shared__ alignas(16) _Float16 Bs[2][4][32 * 32];     // col-major [n][k] per wave
  const int tid  = threadIdx.x;
  const int wave = tid >> 5, lane = tid & 31;
  const int o0 = blockIdx.y * 16;
  const int b  = blockIdx.z;
  const int n0 = (blockIdx.x * 4 + wave) * 32;
  const T* xb = in + (size_t)b * C * SPATIAL;

  const int nsteps = C / 32;
  stage_w16x32(As[0], w, o0, C, 0, tid);
  stage_x32x32<T>(Bs[0][wave], xb + n0, lane);
  __syncthreads();

  v8f acc0 = {}, acc1 = {};
  for (int t = 0; t < nsteps; ++t) {
    const int cur = t & 1;
    if (t + 1 < nsteps) {
      const int kn = (t + 1) * 32;
      __builtin_prefetch(xb + (size_t)kn * SPATIAL + n0, 0, 1);
      stage_w16x32(As[cur ^ 1], w, o0, C, kn, tid);
      stage_x32x32<T>(Bs[cur ^ 1][wave], xb + (size_t)kn * SPATIAL + n0, lane);
    }
    const v16h a  = frag_a(As[cur], 32, lane);
    const v16h b0 = frag_b(&Bs[cur][wave][0], 32, lane);
    const v16h b1 = frag_b(&Bs[cur][wave][16 * 32], 32, lane);
    acc0 = wmma_f16(a, b0, acc0);
    acc1 = wmma_f16(a, b1, acc1);
    __syncthreads();
  }

  const int nc = lane & 15;
  const int mb = (lane >> 4) << 3;
#pragma unroll
  for (int r = 0; r < 8; ++r) {
    const int o = o0 + mb + r;
    const float bo = bias[o];
    float* orow = out + ((size_t)b * O + o) * SPATIAL + n0;
    orow[nc]      = acc0[r] + bo;
    orow[16 + nc] = acc1[r] + bo;
  }
}

// ---------------------------------------------------------------------------
// Depthwise 3x3x3, SAME zero padding.  Writes the f16 mirror used by all
// downstream WMMA stages (halves HBM traffic vs an f32 intermediate).
// ---------------------------------------------------------------------------
__global__ __launch_bounds__(256)
void dwconv3x3x3(const float* __restrict__ in, const float* __restrict__ w,
                 const float* __restrict__ bias, _Float16* __restrict__ out) {
  const size_t idx = (size_t)blockIdx.x * 256 + threadIdx.x;
  const int d = (int)(idx & 31);
  size_t t = idx >> 5;
  const int y = (int)(t & 31); t >>= 5;
  const int z = (int)(t & 31); t >>= 5;
  const int c = (int)(t % (3 * CDIM));
  const int b = (int)(t / (3 * CDIM));
  const float* base = in + ((size_t)b * 3 * CDIM + c) * SPATIAL;
  const float* wt = w + (size_t)c * 27;
  float s = bias[c];
#pragma unroll
  for (int iz = 0; iz < 3; ++iz) {
    const int zz = z + iz - 1;
    if ((unsigned)zz > 31u) continue;
#pragma unroll
    for (int iy = 0; iy < 3; ++iy) {
      const int yy = y + iy - 1;
      if ((unsigned)yy > 31u) continue;
#pragma unroll
      for (int ix = 0; ix < 3; ++ix) {
        const int dd = d + ix - 1;
        if ((unsigned)dd > 31u) continue;
        s += base[((size_t)zz * 32 + yy) * 32 + dd] * wt[(iz * 3 + iy) * 3 + ix];
      }
    }
  }
  out[idx] = (_Float16)s;
}

// ---------------------------------------------------------------------------
// Reciprocal L2 norms of q/k rows from the f16 mirror (f32 accumulation).
// ---------------------------------------------------------------------------
__global__ __launch_bounds__(256)
void rownorm_rsqrt(const _Float16* __restrict__ qkv16, float* __restrict__ scl) {
  __shared__ float red[256];
  const int c = blockIdx.x, b = blockIdx.y, which = blockIdx.z; // 0=q, 1=k
  const v8h* row = (const v8h*)(qkv16 +
      ((size_t)b * 3 * CDIM + which * CDIM + c) * SPATIAL);
  float s = 0.f;
  for (int i = threadIdx.x; i < SPATIAL / 8; i += 256) {
    const v8h v = row[i];
#pragma unroll
    for (int j = 0; j < 8; ++j) { const float f = (float)v[j]; s += f * f; }
  }
  red[threadIdx.x] = s;
  __syncthreads();
  for (int st = 128; st > 0; st >>= 1) {
    if (threadIdx.x < st) red[threadIdx.x] += red[threadIdx.x + st];
    __syncthreads();
  }
  if (threadIdx.x == 0) {
    const float n = sqrtf(red[0]);
    scl[((size_t)which * NB + b) * CDIM + c] = 1.f / fmaxf(n, 1e-12f);
  }
}

// ---------------------------------------------------------------------------
// S[bh,c,e] = q_c . k_e over K=32768.  One block per (b,h); 9 waves cover
// the 3x3 grid of 16x16 tiles.  Triple-buffered pipeline, one barrier per
// K step.  Slabs staged by the Tensor Data Mover (wave 0: Q, wave 1: K),
// tracked with TENSORcnt; fallback is a per-thread b128 copy.
// ---------------------------------------------------------------------------
__global__ __launch_bounds__(288)
void qk_scores_wmma(const _Float16* __restrict__ qkv16, float* __restrict__ S) {
  __shared__ alignas(16) _Float16 Qs[3][48 * 32];   // row-major [m][k]
  __shared__ alignas(16) _Float16 Ks[3][48 * 32];   // col-major [e][k]
  const int bh = blockIdx.x;
  const int b = bh >> 2, h = bh & 3;
  const int tid  = threadIdx.x;
  const int wave = tid >> 5, lane = tid & 31;
  const int mt = wave / 3, nt = wave % 3;
  const _Float16* qbase = qkv16 + ((size_t)b * 3 * CDIM + h * CH) * SPATIAL;
  const _Float16* kbase = qbase + (size_t)CDIM * SPATIAL;

#if HAVE_TDM
  const unsigned qs_lds = (unsigned)(uintptr_t)(&Qs[0][0]);
  const unsigned ks_lds = (unsigned)(uintptr_t)(&Ks[0][0]);
  auto issue = [&](int buf, int k0) {
    if (wave == 0)      tdm_load_48x32_f16(qs_lds + (unsigned)buf * 48 * 32 * 2,
                                           qbase + k0);
    else if (wave == 1) tdm_load_48x32_f16(ks_lds + (unsigned)buf * 48 * 32 * 2,
                                           kbase + k0);
  };
#else
  auto issue = [&](int buf, int k0) {
    for (int i = tid; i < 192; i += 288) {            // 48 rows x 4 chunks
      const int m = i >> 2, c8 = (i & 3) << 3;
      *(v8h*)(&Qs[buf][m * 32 + c8]) =
          *(const v8h*)(qbase + (size_t)m * SPATIAL + k0 + c8);
      *(v8h*)(&Ks[buf][m * 32 + c8]) =
          *(const v8h*)(kbase + (size_t)m * SPATIAL + k0 + c8);
    }
  };
#endif

  const int nsteps = SPATIAL / 32;
  issue(0, 0);
  issue(1, 32);

  v8f acc = {};
  int cur = 0;
  for (int t = 0; t < nsteps; ++t) {
#if HAVE_TDM
    if (wave < 2) {                 // wave-uniform; immediate must be constant
      if (t + 1 < nsteps) __builtin_amdgcn_s_wait_tensorcnt(1);
      else                __builtin_amdgcn_s_wait_tensorcnt(0);
    }
#endif
    __syncthreads();     // slab t visible to all; prior readers of next target done
    const v16h a  = frag_a(&Qs[cur][mt * 16 * 32], 32, lane);
    const v16h bb = frag_b(&Ks[cur][nt * 16 * 32], 32, lane);
    acc = wmma_f16(a, bb, acc);
    if (t + 2 < nsteps) {
      const int nxt = (cur + 2 >= 3) ? cur - 1 : cur + 2;
      issue(nxt, (t + 2) * 32);
    }
    cur = (cur + 1 == 3) ? 0 : cur + 1;
  }

  const int e  = nt * 16 + (lane & 15);
  const int mb = (lane >> 4) << 3;
#pragma unroll
  for (int r = 0; r < 8; ++r) {
    const int m = mt * 16 + mb + r;
    S[((size_t)bh * CH + m) * CH + e] = acc[r];
  }
}

// ---------------------------------------------------------------------------
// attn = softmax_e( S[c,e] * (1/|q_c|) * (1/|k_e|) * temperature[h] )
// ---------------------------------------------------------------------------
__global__ __launch_bounds__(64)
void softmax_rows(const float* __restrict__ S, const float* __restrict__ scl,
                  const float* __restrict__ temp, float* __restrict__ A) {
  const int b = blockIdx.x, h = blockIdx.y, c = threadIdx.x;
  if (c >= CH) return;
  const int bh = b * NHEADS + h;
  const float* row = S + ((size_t)bh * CH + c) * CH;
  const float* sq = scl + (size_t)b * CDIM;              // which=0
  const float* sk = scl + (size_t)(NB + b) * CDIM;       // which=1
  const float t = temp[h] * sq[h * CH + c];
  float vals[CH];
  float mx = -1e30f;
#pragma unroll
  for (int e = 0; e < CH; ++e) {
    const float v = row[e] * t * sk[h * CH + e];
    vals[e] = v;
    mx = fmaxf(mx, v);
  }
  float sum = 0.f;
#pragma unroll
  for (int e = 0; e < CH; ++e) { vals[e] = expf(vals[e] - mx); sum += vals[e]; }
  const float inv = 1.f / sum;
  float* outp = A + ((size_t)bh * CH + c) * CH;
#pragma unroll
  for (int e = 0; e < CH; ++e) outp[e] = vals[e] * inv;
}

// ---------------------------------------------------------------------------
// o1[b, h*CH+c, n] = sum_e attn[bh,c,e] * v[bh,e,n].  K=48 zero-padded to 64
// -> two K=32 WMMA steps.  V read from the f16 mirror; output written f16
// (feeds the templated proj GEMM).
// ---------------------------------------------------------------------------
__global__ __launch_bounds__(128)
void attn_v_wmma(const float* __restrict__ attn, const _Float16* __restrict__ qkv16,
                 _Float16* __restrict__ o1) {
  __shared__ alignas(16) _Float16 Af[16 * 64];         // attn tile, K-padded
  __shared__ alignas(16) _Float16 Bv[4][16 * 32];      // col-major [n][k] per wave
  const int tid  = threadIdx.x;
  const int wave = tid >> 5, lane = tid & 31;
  const int bh = blockIdx.z;
  const int b = bh >> 2, h = bh & 3;
  const int m0 = blockIdx.y * 16;
  const int n0 = (blockIdx.x * 4 + wave) * 16;

  for (int i = tid; i < 16 * 32; i += 128) {           // 512 packed pairs
    const int r = i >> 5, kp = (i & 31) << 1;
    h2 p;
    if (kp < CH) {
      const float2 av = *(const float2*)(attn + ((size_t)bh * CH + m0 + r) * CH + kp);
      p[0] = (_Float16)av.x; p[1] = (_Float16)av.y;
    } else {
      p[0] = (_Float16)0.f; p[1] = (_Float16)0.f;
    }
    *(h2*)(&Af[r * 64 + kp]) = p;
  }
  const _Float16* vbase = qkv16 + ((size_t)b * 3 * CDIM + 2 * CDIM + h * CH) * SPATIAL;

  v8f acc = {};
#pragma unroll
  for (int s = 0; s < 2; ++s) {
    const int k0 = s * 32;
    for (int i = lane; i < 256; i += 32) {             // 16 cols x 16 k-pairs
      const int nn = i & 15, kk = (i >> 4) << 1;
      const int e0 = k0 + kk;
      h2 p;
      p[0] = (e0 < CH)     ? vbase[(size_t)e0 * SPATIAL + n0 + nn] : (_Float16)0.f;
      p[1] = (e0 + 1 < CH) ? vbase[(size_t)(e0 + 1) * SPATIAL + n0 + nn]
                           : (_Float16)0.f;
      *(h2*)(&Bv[wave][nn * 32 + kk]) = p;
    }
    __syncthreads();
    const v16h a  = frag_a(Af + k0, 64, lane);
    const v16h bb = frag_b(Bv[wave], 32, lane);
    acc = wmma_f16(a, bb, acc);
    __syncthreads();
  }

  const int n  = n0 + (lane & 15);
  const int mb = (lane >> 4) << 3;
#pragma unroll
  for (int r = 0; r < 8; ++r) {
    const int c = h * CH + m0 + mb + r;
    o1[((size_t)b * CDIM + c) * SPATIAL + n] = (_Float16)acc[r];
  }
}

// ---------------------------------------------------------------------------
extern "C" void kernel_launch(void* const* d_in, const int* in_sizes, int n_in,
                              void* d_out, int out_size, void* d_ws, size_t ws_size,
                              hipStream_t stream) {
  (void)in_sizes; (void)n_in; (void)out_size; (void)ws_size;
  const float* x      = (const float*)d_in[0];
  const float* qkv_w  = (const float*)d_in[1];
  const float* qkv_b  = (const float*)d_in[2];
  const float* dw_w   = (const float*)d_in[3];
  const float* dw_b   = (const float*)d_in[4];
  const float* proj_w = (const float*)d_in[5];
  const float* proj_b = (const float*)d_in[6];
  const float* temp   = (const float*)d_in[7];
  float* out = (float*)d_out;

  // Workspace (~227 MB): qkv1 f32, qkv16 f16 mirror, small buffers.
  const size_t QKV_ELEMS = (size_t)NB * 3 * CDIM * SPATIAL;   // 37,748,736
  float*    qkv1  = (float*)d_ws;
  _Float16* qkv16 = (_Float16*)(qkv1 + QKV_ELEMS);
  float*    scl   = (float*)(qkv16 + QKV_ELEMS);               // 768 floats
  float*    Sbuf  = scl + 1024;                                // 18432 floats
  float*    Abuf  = Sbuf + (size_t)NB * NHEADS * CH * CH;      // 18432 floats
  _Float16* o1h   = (_Float16*)qkv1;  // qkv1 dead after dwconv; reuse

  // 1) qkv 1x1x1 conv (GEMM, WMMA), f32 input
  conv1x1_wmma<float><<<dim3(SPATIAL / 128, (3 * CDIM) / 16, NB), 128, 0, stream>>>(
      x, qkv_w, qkv_b, qkv1, 3 * CDIM, CDIM);

  // 2) depthwise 3x3x3 -> f16 mirror
  {
    const size_t total = (size_t)NB * 3 * CDIM * SPATIAL;
    dwconv3x3x3<<<dim3((unsigned)(total / 256)), 256, 0, stream>>>(
        qkv1, dw_w, dw_b, qkv16);
  }

  // 3) reciprocal L2 norms of q and k rows
  rownorm_rsqrt<<<dim3(CDIM, NB, 2), 256, 0, stream>>>(qkv16, scl);

  // 4) raw Gram matrices S = Q K^T (WMMA, K = 32768, TDM-staged)
  qk_scores_wmma<<<dim3(NB * NHEADS), 288, 0, stream>>>(qkv16, Sbuf);

  // 5) normalized, temperature-scaled softmax over 48 columns
  softmax_rows<<<dim3(NB, NHEADS), CH, 0, stream>>>(Sbuf, scl, temp, Abuf);

  // 6) out = attn @ V (WMMA, K=48 padded to 64), f16 output
  attn_v_wmma<<<dim3(SPATIAL / 64, CH / 16, NB * NHEADS), 128, 0, stream>>>(
      Abuf, qkv16, o1h);

  // 7) projection 1x1x1 conv (GEMM, WMMA), f16 input, f32 output
  conv1x1_wmma<_Float16><<<dim3(SPATIAL / 128, CDIM / 16, NB), 128, 0, stream>>>(
      o1h, proj_w, proj_b, out, CDIM, CDIM);
}
